// TuckerDecomposedExpert_32375463477352
// MI455X (gfx1250) — compile-verified, matrix operand order checked by reading, not run on hardware
//
#include <hip/hip_runtime.h>

// ---------------------------------------------------------------------------
// TuckerDecomposedExpert on MI455X (gfx1250, wave32, WMMA bf16 16x16x32)
//
//   h  = x @ U_in            [8192,4096]x[4096,256]
//   h2 = h @ Core[expert]    counting-sorted tokens -> batched per-expert GEMM
//   y  = h2 @ U_out          [8192,256]x[256,4096]
//
// Matrix math on v_wmma_f32_16x16x32_bf16. B tiles are staged to LDS with
// CDNA5 async global->LDS copies (ASYNCcnt, inline asm per ISA ch.10/15.18)
// and double-buffered so the copy of tile k+1 overlaps the WMMAs of tile k.
// Weights pre-transposed to N-major so both operand fragments are two
// contiguous b128 loads per lane, matching the ISA 16-bit operand layout
// (lane = m/n mod 16, half = lane/16, frag elems [0..7] = K[half*8 ..],
// elems [8..15] = K[16+half*8 ..]).
// ---------------------------------------------------------------------------

typedef __attribute__((ext_vector_type(16))) __bf16 bf16x16;
typedef __attribute__((ext_vector_type(8)))  __bf16 bf16x8;
typedef __attribute__((ext_vector_type(8)))  float  f32x8;

#define TOKENS   8192
#define IN_DIM   4096
#define OUT_DIM  4096
#define R_IN     256
#define R_OUT    256
#define N_EXP    64

// ---- CDNA5 async global->LDS copy (ASYNCcnt), inline asm -------------------
// VDST = 32-bit LDS byte address (low 32 bits of generic LDS pointer, per ISA
// aperture rule); VADDR = 64-bit global address pair.
__device__ __forceinline__ void async_copy_b128(const __bf16* g, __bf16* l) {
  unsigned loff = (unsigned)(unsigned long long)l;
  asm volatile("global_load_async_to_lds_b128 %0, %1, off"
               :: "v"(loff), "v"((unsigned long long)g) : "memory");
}

__device__ __forceinline__ void wait_async0() {
  asm volatile("s_wait_asynccnt 0x0" ::: "memory");
}

// Load one 16x32 (row-major A) / 32x16 (col-major B) bf16 WMMA fragment for
// this lane from a K-contiguous row pointer. hsel = lane>>4.
__device__ __forceinline__ bf16x16 load_frag(const __bf16* p, int hsel) {
  bf16x8 lo = *(const bf16x8*)(p + hsel * 8);
  bf16x8 hi = *(const bf16x8*)(p + 16 + hsel * 8);
  return __builtin_shufflevector(lo, hi, 0,1,2,3,4,5,6,7,8,9,10,11,12,13,14,15);
}

// ----------------------------- transpose -----------------------------------
// dst[b][c][r] = src[b][r][c]; batch via blockIdx.y. Tiny one-time cost.
__global__ void transpose_bf16(const __bf16* __restrict__ src,
                               __bf16* __restrict__ dst, int R, int C) {
  size_t b = blockIdx.y;
  int i = blockIdx.x * blockDim.x + threadIdx.x;
  if (i >= R * C) return;
  int r = i / C, c = i % C;
  dst[b * (size_t)R * C + (size_t)c * R + r] = src[b * (size_t)R * C + i];
}

// --------------------------- counting sort ---------------------------------
__global__ void zero_counts(int* counts) {
  if (threadIdx.x < N_EXP) counts[threadIdx.x] = 0;
}
__global__ void hist_kernel(const int* __restrict__ idx, int* counts) {
  int t = blockIdx.x * blockDim.x + threadIdx.x;
  if (t < TOKENS) atomicAdd(&counts[idx[t]], 1);
}
__global__ void scan_kernel(const int* __restrict__ counts, int* offs, int* cursor) {
  if (threadIdx.x == 0) {
    int s = 0;
    for (int e = 0; e < N_EXP; ++e) { offs[e] = s; cursor[e] = s; s += counts[e]; }
  }
}
__global__ void scatter_kernel(const int* __restrict__ idx, int* cursor,
                               int* __restrict__ order) {
  int t = blockIdx.x * blockDim.x + threadIdx.x;
  if (t < TOKENS) { int p = atomicAdd(&cursor[idx[t]], 1); order[p] = t; }
}

// ------------------------------ main GEMM ----------------------------------
// C[M,N] = A[M,K] @ B[K,N] with B given transposed (Bt[N,K]). bf16 in/out,
// f32 accumulate. Block = 256 threads = 8 waves; block tile 128x128; each
// wave: 16 rows x 128 cols = 8 WMMA accumulators. Bt tile double-buffered in
// LDS via async copies (stride 40 bf16 -> conflict-free half-wave reads).
#define BM 128
#define BN 128
#define LDSB 40

__global__ __launch_bounds__(256) void gemm_bf16_tn(
    const __bf16* __restrict__ A, const __bf16* __restrict__ Bt,
    __bf16* __restrict__ C, int M, int N, int K) {
  __shared__ __bf16 bs[2][BN * LDSB];          // 2 x 10 KB
  const int lane = threadIdx.x & 31;
  const int wave = threadIdx.x >> 5;
  const int hsel = lane >> 4;
  const int l16  = lane & 15;
  const int rowBlock = blockIdx.x * BM;
  const int colBlock = blockIdx.y * BN;

  // this thread's two 16-byte staging chunks: rows r0 and r0+64, cols c0..c0+7
  const int r0 = threadIdx.x >> 2;             // 0..63
  const int c0 = (threadIdx.x & 3) * 8;        // 0,8,16,24
  const __bf16* brow0 = Bt + (size_t)(colBlock + r0) * K + c0;
  const __bf16* brow1 = Bt + (size_t)(colBlock + r0 + 64) * K + c0;

  const __bf16* arow = A + (size_t)(rowBlock + wave * 16 + l16) * K;

  f32x8 acc[8] = {};

  const int KT = K >> 5;
  // prologue: stage tile 0 into buffer 0
  async_copy_b128(brow0, &bs[0][r0 * LDSB + c0]);
  async_copy_b128(brow1, &bs[0][(r0 + 64) * LDSB + c0]);

  for (int kt = 0; kt < KT; ++kt) {
    wait_async0();                 // own copies (issued last iteration) done
    __syncthreads();               // everyone's copies done / prior reads done

    if (kt + 1 < KT) {             // overlap: stage next tile into alt buffer
      int knext = (kt + 1) << 5;
      __bf16* dst = &bs[(kt + 1) & 1][0];
      async_copy_b128(brow0 + knext, dst + r0 * LDSB + c0);
      async_copy_b128(brow1 + knext, dst + (r0 + 64) * LDSB + c0);
    }

    const __bf16* bsrc = &bs[kt & 1][0];
    bf16x16 afrag = load_frag(arow + (kt << 5), hsel);
    #pragma unroll
    for (int nt = 0; nt < 8; ++nt) {
      bf16x16 bfrag = load_frag(bsrc + (nt * 16 + l16) * LDSB, hsel);
      acc[nt] = __builtin_amdgcn_wmma_f32_16x16x32_bf16(
          false, afrag, false, bfrag, (short)0, acc[nt], false, false);
    }
  }

  // D layout: lane L holds col n=L%16; VGPR v holds row m = (L/16)*8 + v
  #pragma unroll
  for (int nt = 0; nt < 8; ++nt) {
    #pragma unroll
    for (int v = 0; v < 8; ++v) {
      size_t m = rowBlock + wave * 16 + hsel * 8 + v;
      size_t n = colBlock + nt * 16 + l16;
      C[m * N + n] = (__bf16)acc[nt][v];
    }
  }
}

// --------------------------- expert contraction ----------------------------
// h2[tok] = h[tok] @ core[e], tokens grouped by expert. Block = 128 thr =
// 4 waves; block handles one 16-token tile of one expert across all 256
// output cols (each wave 64 cols, 4 accumulators). coreT (8 MB, N-major)
// stays resident in the 192 MB L2.
__global__ __launch_bounds__(128) void expert_mm(
    const __bf16* __restrict__ h, const __bf16* __restrict__ coreT,
    __bf16* __restrict__ h2, const int* __restrict__ order,
    const int* __restrict__ offs, const int* __restrict__ counts) {
  const int e    = blockIdx.x;
  const int tile = blockIdx.y;
  const int cnt  = counts[e];
  if (tile * 16 >= cnt) return;
  const int start = offs[e];
  const int lane = threadIdx.x & 31;
  const int wave = threadIdx.x >> 5;
  const int hsel = lane >> 4;
  const int l16  = lane & 15;

  int rloc = tile * 16 + l16;
  int tokA = order[start + (rloc < cnt ? rloc : 0)];   // dup row 0 on ragged tail
  const __bf16* arow  = h + (size_t)tokA * R_IN;
  const __bf16* bbase = coreT + (size_t)e * R_IN * R_OUT;

  f32x8 acc[4] = {};
  #pragma unroll
  for (int k = 0; k < R_IN; k += 32) {
    bf16x16 afrag = load_frag(arow + k, hsel);
    #pragma unroll
    for (int nt = 0; nt < 4; ++nt) {
      int n = wave * 64 + nt * 16 + l16;
      bf16x16 bfrag = load_frag(bbase + (size_t)n * R_IN + k, hsel);
      acc[nt] = __builtin_amdgcn_wmma_f32_16x16x32_bf16(
          false, afrag, false, bfrag, (short)0, acc[nt], false, false);
    }
  }

  #pragma unroll
  for (int v = 0; v < 8; ++v) {
    int r = tile * 16 + hsel * 8 + v;
    if (r < cnt) {
      int tok = order[start + r];
      #pragma unroll
      for (int nt = 0; nt < 4; ++nt)
        h2[(size_t)tok * R_OUT + wave * 64 + nt * 16 + l16] = (__bf16)acc[nt][v];
    }
  }
}

// ------------------------------- launch ------------------------------------
extern "C" void kernel_launch(void* const* d_in, const int* in_sizes, int n_in,
                              void* d_out, int out_size, void* d_ws, size_t ws_size,
                              hipStream_t stream) {
  const __bf16* x     = (const __bf16*)d_in[0];
  const __bf16* u_in  = (const __bf16*)d_in[1];
  const __bf16* core  = (const __bf16*)d_in[2];
  const __bf16* u_out = (const __bf16*)d_in[3];
  const int* expert_idx = (const int*)d_in[4];
  __bf16* out = (__bf16*)d_out;

  // workspace layout (needs ~20.1 MB)
  char* ws = (char*)d_ws;
  __bf16* u_inT  = (__bf16*)(ws);                       // [R_IN  , IN_DIM]  2 MB
  __bf16* u_outT = (__bf16*)(ws + (2u << 20));          // [OUT_DIM, R_OUT]  2 MB
  __bf16* coreT  = (__bf16*)(ws + (4u << 20));          // [NE, R_OUT, R_IN] 8 MB
  __bf16* h      = (__bf16*)(ws + (12u << 20));         // [T, R_IN]         4 MB
  __bf16* h2     = (__bf16*)(ws + (16u << 20));         // [T, R_OUT]        4 MB
  int* counts = (int*)(ws + (20u << 20));
  int* offs   = counts + 64;
  int* cursor = counts + 128;
  int* order  = counts + 192;                           // [T]

  // 1. transpose weights to N-major
  transpose_bf16<<<dim3((IN_DIM * R_IN + 255) / 256, 1), 256, 0, stream>>>(
      u_in, u_inT, IN_DIM, R_IN);
  transpose_bf16<<<dim3((R_IN * R_OUT + 255) / 256, N_EXP), 256, 0, stream>>>(
      core, coreT, R_IN, R_OUT);
  transpose_bf16<<<dim3((R_OUT * OUT_DIM + 255) / 256, 1), 256, 0, stream>>>(
      u_out, u_outT, R_OUT, OUT_DIM);

  // 2. counting sort of tokens by expert
  zero_counts<<<1, 64, 0, stream>>>(counts);
  hist_kernel<<<TOKENS / 256, 256, 0, stream>>>(expert_idx, counts);
  scan_kernel<<<1, 32, 0, stream>>>(counts, offs, cursor);
  scatter_kernel<<<TOKENS / 256, 256, 0, stream>>>(expert_idx, cursor, order);

  // 3. h = x @ U_in
  gemm_bf16_tn<<<dim3(TOKENS / BM, R_IN / BN), 256, 0, stream>>>(
      x, u_inT, h, TOKENS, R_IN, IN_DIM);

  // 4. h2 = h @ core[expert]  (grouped; idle tiles exit immediately)
  expert_mm<<<dim3(N_EXP, TOKENS / 16), 128, 0, stream>>>(
      h, coreT, h2, order, offs, counts);

  // 5. y = h2 @ U_out
  gemm_bf16_tn<<<dim3(TOKENS / BM, OUT_DIM / BN), 256, 0, stream>>>(
      h2, u_outT, out, TOKENS, OUT_DIM, R_OUT);
}
